// MultiHeadAttention_35716948034077
// MI455X (gfx1250) — compile-verified, matrix operand order checked by reading, not run on hardware
//
#include <hip/hip_runtime.h>
#include <hip/hip_bf16.h>
#include <math.h>

// ---------------------------------------------------------------------------
// MHA forward B=2,S=2048,D=512,H=8,hd=64 on gfx1250 (wave32, WMMA bf16).
// d_out = [out f32 (2*2048*512)] ++ [raw_score f32 (16*2048*2048)]
// ws (bf16 elems): q|k|v [16,2048,64] | heads [2,2048,512] | xb | Wq|Wk|Wv|Wo
// ---------------------------------------------------------------------------

#define DEV __device__ __forceinline__

typedef __attribute__((ext_vector_type(16))) __bf16 v16bf;
typedef __attribute__((ext_vector_type(8)))  __bf16 v8bf;
typedef __attribute__((ext_vector_type(8)))  float  v8f;

static constexpr int   Bb   = 2;
static constexpr int   Ss   = 2048;
static constexpr int   Dd   = 512;
static constexpr int   HB   = 16;             // H*B
static constexpr float NEGF = -4294967295.0f; // -2^32 + 1 (mask fill)

DEV __bf16 f2bf(float f) { return (__bf16)f; }   // native v_cvt_pk_bf16_f32

// 16-element bf16 fragment: this lane's K offsets {0..7} and {16..23} from p.
DEV v16bf frag_bf(const __bf16* p) {
  v8bf lo = *(const v8bf*)(p);
  v8bf hi = *(const v8bf*)(p + 16);
  v16bf f;
#pragma unroll
  for (int i = 0; i < 8; ++i) { f[i] = lo[i]; f[8 + i] = hi[i]; }
  return f;
}

DEV v8f wmma_bf16(v16bf a, v16bf b, v8f c) {
  return __builtin_amdgcn_wmma_f32_16x16x32_bf16(false, a, false, b,
                                                 (short)0, c, false, false);
}

// ---------------------------------------------------------------------------
// Kernel 0: one-shot f32 -> bf16 conversion (8 elems/thread, b128 in/out)
// ---------------------------------------------------------------------------
__global__ void __launch_bounds__(256) k_cvt(const float* __restrict__ s,
                                             __bf16* __restrict__ d, int n8) {
  int i = blockIdx.x * 256 + threadIdx.x;
  if (i >= n8) return;
  const float4* sp = (const float4*)s + (size_t)i * 2;
  float4 a = sp[0], b = sp[1];
  v8bf o;
  o[0] = f2bf(a.x); o[1] = f2bf(a.y); o[2] = f2bf(a.z); o[3] = f2bf(a.w);
  o[4] = f2bf(b.x); o[5] = f2bf(b.y); o[6] = f2bf(b.z); o[7] = f2bf(b.w);
  *(v8bf*)(d + (size_t)i * 8) = o;
}

// ---------------------------------------------------------------------------
// Kernel 1: fused Q/K/V projections, 16x64 tile per wave (one head's cols).
// A fragment reused across 4 B columns: 1 A-load + 4 B-loads per 4 WMMAs.
// ---------------------------------------------------------------------------
__global__ void __launch_bounds__(256) k_qkv(
    const __bf16* __restrict__ xb,
    const __bf16* __restrict__ wq, const float* __restrict__ bq,
    const __bf16* __restrict__ wk, const float* __restrict__ bk,
    const __bf16* __restrict__ wv, const float* __restrict__ bv,
    __bf16* __restrict__ qkv)
{
  int gwave = (blockIdx.x * 256 + threadIdx.x) >> 5;
  int lane  = threadIdx.x & 31;
  int proj  = gwave / (256 * 8);
  int rem   = gwave % (256 * 8);
  int n0    = (rem >> 3) << 4;   // over B*S = 4096
  int o0    = (rem & 7) << 6;    // over H*hd = 512, 64-wide (one head)

  const __bf16* W    = (proj == 0) ? wq : (proj == 1) ? wk : wv;
  const float*  bias = (proj == 0) ? bq : (proj == 1) ? bk : bv;
  __bf16* outp = qkv + (size_t)proj * ((size_t)HB * Ss * 64);

  int li = lane & 15, kb = (lane < 16) ? 0 : 8, rofs = (lane < 16) ? 0 : 8;
  const __bf16* xrow = xb + (size_t)(n0 + li) * Dd;
  const __bf16* w0 = W + (size_t)(o0 + li) * Dd;
  const __bf16* w1 = w0 + (size_t)16 * Dd;
  const __bf16* w2 = w0 + (size_t)32 * Dd;
  const __bf16* w3 = w0 + (size_t)48 * Dd;

  v8f c0 = {}, c1 = {}, c2 = {}, c3 = {};
#pragma unroll 2
  for (int d0 = 0; d0 < Dd; d0 += 32) {
    v16bf a = frag_bf(xrow + d0 + kb);
    c0 = wmma_bf16(a, frag_bf(w0 + d0 + kb), c0);
    c1 = wmma_bf16(a, frag_bf(w1 + d0 + kb), c1);
    c2 = wmma_bf16(a, frag_bf(w2 + d0 + kb), c2);
    c3 = wmma_bf16(a, frag_bf(w3 + d0 + kb), c3);
  }

  int h = o0 >> 6;
  float b0_ = bias[o0 + li], b1_ = bias[o0 + 16 + li];
  float b2_ = bias[o0 + 32 + li], b3_ = bias[o0 + 48 + li];
#pragma unroll
  for (int r = 0; r < 8; ++r) {
    int n = n0 + rofs + r;
    int bb = n >> 11, s = n & 2047;
    __bf16* op = outp + ((size_t)(h * Bb + bb) * Ss + s) * 64 + li;
    op[0]  = f2bf(c0[r] + b0_);
    op[16] = f2bf(c1[r] + b1_);
    op[32] = f2bf(c2[r] + b2_);
    op[48] = f2bf(c3[r] + b3_);
  }
}

// ---------------------------------------------------------------------------
// Kernel 2: flash attention. Block = 8 waves sharing one (h,b); K tile staged
// to LDS via async-to-LDS DMA, V tile staged transposed; per-wave 16-row
// online softmax; full raw-score matrix streamed with non-temporal stores.
// ---------------------------------------------------------------------------
__global__ void __launch_bounds__(256) k_attn(
    const __bf16* __restrict__ ws,
    __bf16* __restrict__ heads,
    float* __restrict__ score)
{
  __shared__ __bf16 kt[32 * 72];   // [t_loc][qd], padded stride (bank-safe)
  __shared__ __bf16 vt[64 * 40];   // [vcol][t_loc], padded stride
  __shared__ __bf16 plds[8 * 512]; // per-wave 16x32 P tile

  const __bf16* qws = ws;
  const __bf16* kws = ws + (size_t)HB * Ss * 64;
  const __bf16* vws = ws + (size_t)2 * HB * Ss * 64;

  int tid = threadIdx.x, lane = tid & 31, wl = tid >> 5;
  int hb   = blockIdx.x >> 4;          // h*B + b
  int m0b  = (blockIdx.x & 15) << 7;   // 128-row span per block
  int m0   = m0b + (wl << 4);          // this wave's 16-row block
  int bmax = m0b + 127;
  int li = lane & 15, kb = (lane < 16) ? 0 : 8, rofs = (lane < 16) ? 0 : 8;

  // staging roles: thread -> (row 0..31, 8-col group)
  int srow = tid >> 3;
  int scg  = (tid & 7) << 3;
  unsigned k_lds = (unsigned)(uintptr_t)&kt[srow * 72 + scg]; // LDS byte offs
  const __bf16* kgbase = kws + ((size_t)hb * Ss + srow) * 64 + scg;
  const __bf16* vgbase = vws + ((size_t)hb * Ss + srow) * 64 + scg;

  __bf16* myp = plds + wl * 512;

  const __bf16* qrow = qws + ((size_t)hb * Ss + m0 + li) * 64;
  v16bf qa0 = frag_bf(qrow + kb);
  v16bf qa1 = frag_bf(qrow + 32 + kb);

  v8f acc0 = {}, acc1 = {}, acc2 = {}, acc3 = {};
  float rmax[8], rsum[8];
#pragma unroll
  for (int r = 0; r < 8; ++r) { rmax[r] = -INFINITY; rsum[r] = 0.f; }

  for (int t0 = 0; t0 < Ss; t0 += 32) {
    __syncthreads();   // protect LDS tiles from previous iteration's readers

    // K tile 32x64 -> LDS via async DMA (per-lane 16B chunks)
    unsigned long long kga =
        (unsigned long long)(uintptr_t)(kgbase + (size_t)t0 * 64);
    asm volatile("global_load_async_to_lds_b128 %0, %1, off"
                 :: "v"(k_lds), "v"(kga) : "memory");

    // V tile 32x64 -> LDS transposed (coalesced b128 in, b16 scatter)
    if (t0 <= bmax) {
      v8bf vv = *(const v8bf*)(vgbase + (size_t)t0 * 64);
#pragma unroll
      for (int i = 0; i < 8; ++i) vt[(scg + i) * 40 + srow] = vv[i];
    }
    asm volatile("s_wait_asynccnt 0x0" ::: "memory");
    __syncthreads();

    // ---- QK^T (K=64 -> 2 WMMA per 16-col fragment)
    v8f c0 = {}, c1 = {};
    c0 = wmma_bf16(qa0, frag_bf(&kt[li * 72 + kb]), c0);
    c0 = wmma_bf16(qa1, frag_bf(&kt[li * 72 + 32 + kb]), c0);
    c1 = wmma_bf16(qa0, frag_bf(&kt[(16 + li) * 72 + kb]), c1);
    c1 = wmma_bf16(qa1, frag_bf(&kt[(16 + li) * 72 + 32 + kb]), c1);
#pragma unroll
    for (int r = 0; r < 8; ++r) { c0[r] *= 0.125f; c1[r] *= 0.125f; }

    // ---- raw score write (pre-mask), streaming
    float* srowp = score + ((size_t)hb * Ss + m0 + rofs) * Ss + t0 + li;
#pragma unroll
    for (int r = 0; r < 8; ++r) {
      __builtin_nontemporal_store(c0[r], srowp + (size_t)r * Ss);
      __builtin_nontemporal_store(c1[r], srowp + (size_t)r * Ss + 16);
    }

    if (t0 <= m0 + 15) {   // wave-uniform: tile intersects causal region
      // causal mask
#pragma unroll
      for (int r = 0; r < 8; ++r) {
        int row = m0 + rofs + r;
        if (t0 + li > row)      c0[r] = NEGF;
        if (t0 + 16 + li > row) c1[r] = NEGF;
      }
      // online softmax (row stats across 16-lane halves)
      float corr[8];
#pragma unroll
      for (int r = 0; r < 8; ++r) {
        float tm = fmaxf(c0[r], c1[r]);
        tm = fmaxf(tm, __shfl_xor(tm, 1, 32));
        tm = fmaxf(tm, __shfl_xor(tm, 2, 32));
        tm = fmaxf(tm, __shfl_xor(tm, 4, 32));
        tm = fmaxf(tm, __shfl_xor(tm, 8, 32));
        float nm = fmaxf(rmax[r], tm);
        corr[r] = __expf(rmax[r] - nm);
        rmax[r] = nm;
        float p0 = __expf(c0[r] - nm);
        float p1 = __expf(c1[r] - nm);
        c0[r] = p0; c1[r] = p1;
        float rs = p0 + p1;
        rs += __shfl_xor(rs, 1, 32);
        rs += __shfl_xor(rs, 2, 32);
        rs += __shfl_xor(rs, 4, 32);
        rs += __shfl_xor(rs, 8, 32);
        rsum[r] = rsum[r] * corr[r] + rs;
      }
#pragma unroll
      for (int r = 0; r < 8; ++r) {
        acc0[r] *= corr[r]; acc1[r] *= corr[r];
        acc2[r] *= corr[r]; acc3[r] *= corr[r];
      }
      // relayout P: C-frag -> LDS -> A-frag (intra-wave)
#pragma unroll
      for (int r = 0; r < 8; ++r) {
        myp[(rofs + r) * 32 + li]      = f2bf(c0[r]);
        myp[(rofs + r) * 32 + 16 + li] = f2bf(c1[r]);
      }
      asm volatile("s_wait_dscnt 0x0" ::: "memory");
      v16bf pa = frag_bf(myp + li * 32 + kb);

      // PV from transposed V tile in LDS
      acc0 = wmma_bf16(pa, frag_bf(&vt[(0  + li) * 40 + kb]), acc0);
      acc1 = wmma_bf16(pa, frag_bf(&vt[(16 + li) * 40 + kb]), acc1);
      acc2 = wmma_bf16(pa, frag_bf(&vt[(32 + li) * 40 + kb]), acc2);
      acc3 = wmma_bf16(pa, frag_bf(&vt[(48 + li) * 40 + kb]), acc3);
    }
  }

  // finalize: normalize, store heads[b,s,h*64+v] bf16
  int h = hb / Bb, bb = hb % Bb;
#pragma unroll
  for (int r = 0; r < 8; ++r) {
    float inv = 1.0f / rsum[r];
    size_t off = ((size_t)bb * Ss + m0 + rofs + r) * (size_t)Dd + h * 64 + li;
    heads[off +  0] = f2bf(acc0[r] * inv);
    heads[off + 16] = f2bf(acc1[r] * inv);
    heads[off + 32] = f2bf(acc2[r] * inv);
    heads[off + 48] = f2bf(acc3[r] * inv);
  }
}

// ---------------------------------------------------------------------------
// Kernel 3: out = heads @ Wo^T + bo (f32), 16x64 tile per wave.
// ---------------------------------------------------------------------------
__global__ void __launch_bounds__(256) k_outproj(
    const __bf16* __restrict__ heads,
    const __bf16* __restrict__ wo, const float* __restrict__ bo,
    float* __restrict__ out)
{
  int gwave = (blockIdx.x * 256 + threadIdx.x) >> 5;
  int lane  = threadIdx.x & 31;
  int n0 = (gwave >> 3) << 4;
  int o0 = (gwave & 7) << 6;
  int li = lane & 15, kb = (lane < 16) ? 0 : 8, rofs = (lane < 16) ? 0 : 8;

  const __bf16* arow = heads + (size_t)(n0 + li) * Dd;
  const __bf16* w0 = wo + (size_t)(o0 + li) * Dd;
  const __bf16* w1 = w0 + (size_t)16 * Dd;
  const __bf16* w2 = w0 + (size_t)32 * Dd;
  const __bf16* w3 = w0 + (size_t)48 * Dd;

  v8f c0 = {}, c1 = {}, c2 = {}, c3 = {};
#pragma unroll 2
  for (int d0 = 0; d0 < Dd; d0 += 32) {
    v16bf a = frag_bf(arow + d0 + kb);
    c0 = wmma_bf16(a, frag_bf(w0 + d0 + kb), c0);
    c1 = wmma_bf16(a, frag_bf(w1 + d0 + kb), c1);
    c2 = wmma_bf16(a, frag_bf(w2 + d0 + kb), c2);
    c3 = wmma_bf16(a, frag_bf(w3 + d0 + kb), c3);
  }

  float b0_ = bo[o0 + li], b1_ = bo[o0 + 16 + li];
  float b2_ = bo[o0 + 32 + li], b3_ = bo[o0 + 48 + li];
#pragma unroll
  for (int r = 0; r < 8; ++r) {
    int n = n0 + rofs + r;
    float* op = out + (size_t)n * Dd + o0 + li;
    op[0]  = c0[r] + b0_;
    op[16] = c1[r] + b1_;
    op[32] = c2[r] + b2_;
    op[48] = c3[r] + b3_;
  }
}

// ---------------------------------------------------------------------------
extern "C" void kernel_launch(void* const* d_in, const int* in_sizes, int n_in,
                              void* d_out, int out_size, void* d_ws, size_t ws_size,
                              hipStream_t stream) {
  const float* x  = (const float*)d_in[0];
  const float* Wq = (const float*)d_in[1];
  const float* bq = (const float*)d_in[2];
  const float* Wk = (const float*)d_in[3];
  const float* bk = (const float*)d_in[4];
  const float* Wv = (const float*)d_in[5];
  const float* bv = (const float*)d_in[6];
  const float* Wo = (const float*)d_in[7];
  const float* bo = (const float*)d_in[8];

  float* out   = (float*)d_out;                  // [B,S,D]
  float* score = out + (size_t)Bb * Ss * Dd;     // [H*B,S,S]

  __bf16* ws    = (__bf16*)d_ws;
  __bf16* qkv   = ws;                                    // 3 * 2,097,152
  __bf16* heads = ws + (size_t)3 * HB * Ss * 64;         // 2,097,152
  __bf16* xb    = heads + (size_t)Bb * Ss * Dd;          // 2,097,152
  __bf16* wqb   = xb + (size_t)Bb * Ss * Dd;             // 262,144 each
  __bf16* wkb   = wqb + (size_t)Dd * Dd;
  __bf16* wvb   = wkb + (size_t)Dd * Dd;
  __bf16* wob   = wvb + (size_t)Dd * Dd;

  // one-shot conversions (memory-bound, tiny)
  k_cvt<<<1024, 256, 0, stream>>>(x,  xb,  (Bb * Ss * Dd) / 8);
  k_cvt<<<128,  256, 0, stream>>>(Wq, wqb, (Dd * Dd) / 8);
  k_cvt<<<128,  256, 0, stream>>>(Wk, wkb, (Dd * Dd) / 8);
  k_cvt<<<128,  256, 0, stream>>>(Wv, wvb, (Dd * Dd) / 8);
  k_cvt<<<128,  256, 0, stream>>>(Wo, wob, (Dd * Dd) / 8);

  // 3 proj * 256 row-tiles * 8 col-tiles(64-wide) = 6144 waves / 8 per block
  k_qkv<<<768, 256, 0, stream>>>(xb, wqb, bq, wkb, bk, wvb, bv, qkv);
  // 16 (h,b) * 16 blocks of 8 row-tiles
  k_attn<<<256, 256, 0, stream>>>(qkv, heads, score);
  // 256 row-tiles * 8 col-tiles = 2048 waves / 8 per block
  k_outproj<<<256, 256, 0, stream>>>(heads, wob, bo, out);
}
